// HaloVLM_38981123178959
// MI455X (gfx1250) — matrix operand order repaired and checked by reading, not compiled, not run on hardware
//
#include <hip/hip_runtime.h>
#include <hip/hip_bf16.h>
#include <math.h>

// ---------------------------------------------------------------------------
// HaloVLM forward for MI455X (gfx1250, wave32, WMMA).
// GEMMs: v_wmma_f32_16x16x32_bf16, 64x128 workgroup tile, double-buffered
//        GLOBAL_LOAD_ASYNC_TO_LDS_B128 weight/activation streaming (ASYNCcnt).
// Attention: flash-style WMMA for QK^T and PV, online softmax, wave32 shfl.
// Weight-bandwidth bound => activations live in an L2-sized workspace and
// every weight streams from HBM exactly once, overlapped with WMMA.
// ---------------------------------------------------------------------------

typedef __attribute__((ext_vector_type(16))) __bf16 v16bf;
typedef __attribute__((ext_vector_type(8)))  float  v8f;

#define DMODEL 768
#define FFDIM  3072
#define TDEC   1416        // 392 prepended + 1024 text
#define TVIT   196
#define NPRE   392
#define VOCAB  32000
#define SEQ    1024
#define BATCH  2
#define NHEAD  12
#define HDIM   64
#define IMAGE_TOK  31999
#define STATE_TOK  31998
#define ACTION_TOK 31997

static __device__ __forceinline__ v8f wmma_bf16(v16bf a, v16bf b, v8f c) {
  return __builtin_amdgcn_wmma_f32_16x16x32_bf16(false, a, false, b, (short)0, c,
                                                 false, false);
}

static __device__ __forceinline__ float gelu_exact(float x) {
  return 0.5f * x * (1.0f + erff(x * 0.70710678118654752f));
}

// CDNA5 async global->LDS copy (GV mode), tracked by ASYNCcnt.
// vdst = LDS byte address, vaddr = 64-bit global address (16B per lane).
static __device__ __forceinline__ void async_b128(unsigned lds_addr,
                                                  const float* gp) {
  asm volatile("global_load_async_to_lds_b128 %0, %1, off"
               :: "v"(lds_addr), "v"((unsigned long long)(size_t)gp)
               : "memory");
}
// Async loads complete in order: waiting <=6 retires everything older than the
// newest 6-instruction tile batch.
static __device__ __forceinline__ void wait_async_prev() {
  asm volatile("s_wait_asynccnt 0x6" ::: "memory");
}
static __device__ __forceinline__ void wait_async_all() {
  asm volatile("s_wait_asynccnt 0x0" ::: "memory");
}

// ---------------------------------------------------------------------------
// Tiled WMMA GEMM:  C[M,N] = epilogue( A[M,K] @ W[K,N] + bias )
//   epilogue: +aux[(row%auxmod)*N+col] (vit pos) | GELU | +res[row*N+col]
// 64x128x32 tile, 256 threads = 8 waves, each wave -> 32x32 (4 WMMAs/step).
// fp32 tiles land in LDS via async DMA (double buffered), bf16 conversion at
// fragment build. Requires K%32==0 and N%128==0 (all WMMA GEMMs here comply).
// ---------------------------------------------------------------------------
__global__ __launch_bounds__(256)
void gemm_wmma(const float* __restrict__ A, const float* __restrict__ W,
               const float* __restrict__ bias, float* __restrict__ C,
               const float* __restrict__ res, const float* __restrict__ aux,
               int M, int N, int K, int auxmod, int fuse_gelu) {
  __shared__ float sA32[2][64][32];    // 16 KB
  __shared__ float sB32[2][32][128];   // 32 KB

  const int tid  = threadIdx.x;
  const int lane = tid & 31;
  const int wv   = tid >> 5;           // 0..7
  const int m0   = blockIdx.y * 64;
  const int n0   = blockIdx.x * 128;
  const int rb0  = (wv >> 2) * 32;     // 0 / 32
  const int c0   = (wv & 3) * 32;      // 0,32,64,96
  const int lm   = lane & 15;
  const int hi   = lane >> 4;

  // 6 async ops per thread per tile (2 for A, 4 for B); per-wave ASYNCcnt +=6.
  auto issue_tile = [&](int k0, int bsel) {
#pragma unroll
    for (int j = 0; j < 2; ++j) {                    // A: 64x32 fp32
      const int chunk = tid + 256 * j;
      const int r  = chunk >> 3;
      const int c4 = (chunk & 7) * 4;
      int row = m0 + r; if (row >= M) row = M - 1;   // clamp, keep EXEC full
      async_b128((unsigned)(size_t)&sA32[bsel][r][c4],
                 A + (size_t)row * K + k0 + c4);
    }
#pragma unroll
    for (int j = 0; j < 4; ++j) {                    // B: 32x128 fp32
      const int chunk = tid + 256 * j;
      const int r  = chunk >> 5;
      const int c4 = (chunk & 31) * 4;
      async_b128((unsigned)(size_t)&sB32[bsel][r][c4],
                 W + (size_t)(k0 + r) * N + n0 + c4);
      if (k0 + 64 < K)   // speculative prefetch of the tile after next
        __builtin_prefetch(W + (size_t)(k0 + 64 + r) * N + n0 + c4, 0, 1);
    }
  };

  v8f acc00 = {}, acc01 = {}, acc10 = {}, acc11 = {};

  int buf = 0;
  issue_tile(0, 0);
  for (int k0 = 0; k0 < K; k0 += 32) {
    if (k0 + 32 < K) { issue_tile(k0 + 32, buf ^ 1); wait_async_prev(); }
    else             { wait_async_all(); }
    __syncthreads();   // tile `buf` visible to all waves

    // fragments (layouts per CDNA5 ISA §7.12.2), fp32->bf16 here
    v16bf a0, a1, b0, b1;
    const float* Ar0 = &sA32[buf][rb0 + lm][0];
    const float* Ar1 = &sA32[buf][rb0 + 16 + lm][0];
#pragma unroll
    for (int i = 0; i < 16; ++i) {
      const int kk = ((i & 8) << 1) + hi * 8 + (i & 7);   // A K-index per lane
      a0[i] = (__bf16)Ar0[kk];
      a1[i] = (__bf16)Ar1[kk];
      const int kb = ((i & 8) << 1) + hi * 8 + (i & 7);   // B K-index per lane
      b0[i] = (__bf16)sB32[buf][kb][c0 + lm];
      b1[i] = (__bf16)sB32[buf][kb][c0 + 16 + lm];
    }
    acc00 = wmma_bf16(a0, b0, acc00);
    acc01 = wmma_bf16(a0, b1, acc01);
    acc10 = wmma_bf16(a1, b0, acc10);
    acc11 = wmma_bf16(a1, b1, acc11);

    __syncthreads();   // everyone done reading `buf` before it is re-filled
    buf ^= 1;
  }

  // ---- epilogue: 4 C tiles per wave ----
#pragma unroll
  for (int t = 0; t < 4; ++t) {
    v8f acc = (t == 0) ? acc00 : (t == 1) ? acc01 : (t == 2) ? acc10 : acc11;
    const int rbase = m0 + rb0 + (t >> 1) * 16 + hi * 8;
    const int col   = n0 + c0 + (t & 1) * 16 + lm;
#pragma unroll
    for (int r = 0; r < 8; ++r) {
      const int row = rbase + r;
      if (row < M) {
        float x = acc[r] + bias[col];
        if (aux) x += aux[(size_t)(row % auxmod) * N + col];
        if (fuse_gelu) x = gelu_exact(x);
        if (res) x += res[(size_t)row * N + col];
        C[(size_t)row * N + col] = x;
      }
    }
  }
}

// ---------------------------------------------------------------------------
// Flash attention, one wave (32 lanes) per (batch, head, 16-query tile).
// Q/K/V/O are [rows, 768] with head offset h*64. WMMA for QK^T and PV.
// ---------------------------------------------------------------------------
__global__ __launch_bounds__(32)
void attn_wmma(const float* __restrict__ Qb, const float* __restrict__ Kb,
               const float* __restrict__ Vb, float* __restrict__ Ob,
               int T, int causal) {
  __shared__ __bf16 sP[16][32];

  const int lane = threadIdx.x;
  const int lm = lane & 15;
  const int hi = lane >> 4;
  const int q0 = blockIdx.x * 16;
  const int h  = blockIdx.y;
  const int b  = blockIdx.z;
  const size_t base = (size_t)b * T * DMODEL + (size_t)h * HDIM;

  v16bf aq0, aq1;
  {
    int rq = q0 + lm; if (rq >= T) rq = T - 1;
    const float* qp = Qb + base + (size_t)rq * DMODEL;
#pragma unroll
    for (int i = 0; i < 16; ++i) {
      const int kk = ((i & 8) << 1) + hi * 8 + (i & 7);
      aq0[i] = (__bf16)qp[kk];
      aq1[i] = (__bf16)qp[32 + kk];
    }
  }

  float mrow[8], lrow[8];
  v8f o0 = {}, o1 = {}, o2 = {}, o3 = {};
#pragma unroll
  for (int r = 0; r < 8; ++r) { mrow[r] = -3.0e38f; lrow[r] = 0.f; }

  const int jend = causal ? ((q0 + 16 < T) ? q0 + 16 : T) : T;
  for (int j0 = 0; j0 < jend; j0 += 32) {
    v16bf bk00, bk01, bk10, bk11;
    {
      const int key0 = j0 + lm;
      const int key1 = j0 + 16 + lm;
      const float* kp0 = Kb + base + (size_t)((key0 < T) ? key0 : 0) * DMODEL;
      const float* kp1 = Kb + base + (size_t)((key1 < T) ? key1 : 0) * DMODEL;
      const float z0 = (key0 < T) ? 1.f : 0.f;
      const float z1 = (key1 < T) ? 1.f : 0.f;
#pragma unroll
      for (int i = 0; i < 16; ++i) {
        const int kk = ((i & 8) << 1) + hi * 8 + (i & 7);
        bk00[i] = (__bf16)(kp0[kk] * z0);
        bk01[i] = (__bf16)(kp0[32 + kk] * z0);
        bk10[i] = (__bf16)(kp1[kk] * z1);
        bk11[i] = (__bf16)(kp1[32 + kk] * z1);
      }
    }
    v8f s0 = {}, s1 = {};
    s0 = wmma_bf16(aq0, bk00, s0);
    s0 = wmma_bf16(aq1, bk01, s0);
    s1 = wmma_bf16(aq0, bk10, s1);
    s1 = wmma_bf16(aq1, bk11, s1);

#pragma unroll
    for (int r = 0; r < 8; ++r) {
      const int row = q0 + r + hi * 8;
      const int cA = j0 + lm;
      const int cB = j0 + 16 + lm;
      float x0 = s0[r] * 0.125f;   // 1/sqrt(64)
      float x1 = s1[r] * 0.125f;
      if (cA >= T || (causal && cA > row)) x0 = -3.0e38f;
      if (cB >= T || (causal && cB > row)) x1 = -3.0e38f;
      float cur = fmaxf(x0, x1);
      cur = fmaxf(cur, __shfl_xor(cur, 1, 32));
      cur = fmaxf(cur, __shfl_xor(cur, 2, 32));
      cur = fmaxf(cur, __shfl_xor(cur, 4, 32));
      cur = fmaxf(cur, __shfl_xor(cur, 8, 32));
      const float mn = fmaxf(mrow[r], cur);
      const float alpha = (mn <= -1.0e37f) ? 1.0f : expf(mrow[r] - mn);
      const float p0 = expf(x0 - mn);
      const float p1 = expf(x1 - mn);
      float sum = p0 + p1;
      sum += __shfl_xor(sum, 1, 32);
      sum += __shfl_xor(sum, 2, 32);
      sum += __shfl_xor(sum, 4, 32);
      sum += __shfl_xor(sum, 8, 32);
      lrow[r] = lrow[r] * alpha + sum;
      mrow[r] = mn;
      o0[r] = o0[r] * alpha; o1[r] = o1[r] * alpha;
      o2[r] = o2[r] * alpha; o3[r] = o3[r] * alpha;
      sP[r + hi * 8][lm]      = (__bf16)p0;
      sP[r + hi * 8][16 + lm] = (__bf16)p1;
    }
    __syncthreads();

    v16bf ap;
#pragma unroll
    for (int i = 0; i < 16; ++i) {
      const int kk = hi * 16 + i;
      ap[i] = sP[lm][kk];
    }
    __syncthreads();

#pragma unroll
    for (int c = 0; c < 4; ++c) {
      v16bf bv;
#pragma unroll
      for (int i = 0; i < 16; ++i) {
        const int key = j0 + hi * 16 + i;
        float val = 0.f;
        if (key < T) val = Vb[base + (size_t)key * DMODEL + c * 16 + lm];
        bv[i] = (__bf16)val;
      }
      if (c == 0)      o0 = wmma_bf16(ap, bv, o0);
      else if (c == 1) o1 = wmma_bf16(ap, bv, o1);
      else if (c == 2) o2 = wmma_bf16(ap, bv, o2);
      else             o3 = wmma_bf16(ap, bv, o3);
    }
  }

#pragma unroll
  for (int r = 0; r < 8; ++r) {
    const int row = q0 + r + hi * 8;
    if (row < T) {
      const float inv = (lrow[r] > 0.f) ? 1.0f / lrow[r] : 0.f;
      float* op = Ob + base + (size_t)row * DMODEL;
      op[lm]      = o0[r] * inv;
      op[16 + lm] = o1[r] * inv;
      op[32 + lm] = o2[r] * inv;
      op[48 + lm] = o3[r] * inv;
    }
  }
}

// ---------------------------------------------------------------------------
// LayerNorm: one wave per 768-wide row, wave32 shfl reductions.
// ---------------------------------------------------------------------------
__global__ __launch_bounds__(32)
void layernorm_row(const float* __restrict__ X, float* __restrict__ Y,
                   const float* __restrict__ g, const float* __restrict__ bt,
                   int N) {
  const int row = blockIdx.x;
  const float* x = X + (size_t)row * N;
  float* y = Y + (size_t)row * N;
  const int lane = threadIdx.x;

  float s = 0.f;
  for (int i = lane; i < N; i += 32) s += x[i];
  for (int o = 16; o > 0; o >>= 1) s += __shfl_xor(s, o, 32);
  const float mean = s / (float)N;

  float v = 0.f;
  for (int i = lane; i < N; i += 32) { const float d = x[i] - mean; v += d * d; }
  for (int o = 16; o > 0; o >>= 1) v += __shfl_xor(v, o, 32);
  const float rstd = rsqrtf(v / (float)N + 1e-5f);

  for (int i = lane; i < N; i += 32)
    y[i] = (x[i] - mean) * rstd * g[i] + bt[i];
}

// --------------------------- small helper kernels ---------------------------

__global__ void patchify(const float* __restrict__ images, float* __restrict__ XP) {
  const int idx = blockIdx.x * blockDim.x + threadIdx.x;
  if (idx >= 784 * 768) return;
  const int c  = idx % 768;
  const int rp = idx / 768;           // [0,784): (b*2+n)*196 + p
  const int bi = rp / 196;
  const int p  = rp % 196;
  const int gy = p / 14, gx = p % 14;
  const int ch = c / 256;
  const int py = (c % 256) / 16, px = c % 16;
  const int yy = gy * 16 + py, xx = gx * 16 + px;
  XP[idx] = images[(((size_t)bi * 3 + ch) * 224 + yy) * 224 + xx];
}

__global__ void small_gemm(const float* __restrict__ A, const float* __restrict__ W,
                           const float* __restrict__ bias, float* __restrict__ C,
                           int M, int N, int K, int fuse_gelu) {
  const int idx = blockIdx.x * blockDim.x + threadIdx.x;
  if (idx >= M * N) return;
  const int m = idx / N, n = idx % N;
  float s = bias[n];
  for (int k = 0; k < K; ++k) s += A[(size_t)m * K + k] * W[(size_t)k * N + n];
  if (fuse_gelu) s = gelu_exact(s);
  C[idx] = s;
}

__global__ void find_positions(const int* __restrict__ ids,
                               int* __restrict__ pos_state,
                               int* __restrict__ pos_act) {
  const int b = threadIdx.x;
  if (b >= BATCH) return;
  for (int j = 0; j < 2; ++j) pos_state[b * 2 + j] = 0;
  for (int j = 0; j < 4; ++j) pos_act[b * 4 + j] = 0;
  int cs = 0, ca = 0;
  for (int s = 0; s < SEQ; ++s) {
    const int id = ids[b * SEQ + s];
    if (id == STATE_TOK  && cs < 2) pos_state[b * 2 + cs++] = s;
    if (id == ACTION_TOK && ca < 4) pos_act[b * 4 + ca++]   = s;
  }
}

__global__ void embed_text(const int* __restrict__ ids,
                           const float* __restrict__ tok_emb,
                           const float* __restrict__ pos_emb,
                           float* __restrict__ H) {
  const int bs = blockIdx.x;
  const int b = bs / SEQ, s = bs % SEQ;
  const int id = ids[bs];
  const bool special = (id == IMAGE_TOK) || (id == STATE_TOK);
  const int t = NPRE + s;
  float* dst = H + ((size_t)b * TDEC + t) * DMODEL;
  const float* pe = pos_emb + (size_t)t * DMODEL;
  const float* te = tok_emb + (size_t)id * DMODEL;
  for (int c = threadIdx.x; c < DMODEL; c += blockDim.x)
    dst[c] = (special ? 0.f : te[c]) + pe[c];
}

__global__ void assemble_img(const float* __restrict__ img_proj,
                             const float* __restrict__ pos_emb,
                             float* __restrict__ H) {
  const int i = blockIdx.x;               // [0,784)
  const int b = i / NPRE, t = i % NPRE;
  const float* src = img_proj + (size_t)i * DMODEL;
  const float* pe  = pos_emb + (size_t)t * DMODEL;
  float* dst = H + ((size_t)b * TDEC + t) * DMODEL;
  for (int c = threadIdx.x; c < DMODEL; c += blockDim.x)
    dst[c] = src[c] + pe[c];
}

__global__ void scatter_state(const float* __restrict__ SE,
                              const float* __restrict__ pos_emb,
                              const int* __restrict__ pos_state,
                              float* __restrict__ H) {
  const int i = blockIdx.x;               // [0, B*2)
  const int b = i / 2;
  const int t = NPRE + pos_state[i];
  float* dst = H + ((size_t)b * TDEC + t) * DMODEL;
  const float* pe  = pos_emb + (size_t)t * DMODEL;
  const float* src = SE + (size_t)i * DMODEL;
  for (int c = threadIdx.x; c < DMODEL; c += blockDim.x)
    dst[c] = src[c] + pe[c];
}

__global__ void gather_hid(const float* __restrict__ HN,
                           const int* __restrict__ pos_act,
                           float* __restrict__ HID) {
  const int i = blockIdx.x;               // [0, B*4)
  const int b = i / 4;
  const int t = NPRE + pos_act[i];
  const float* src = HN + ((size_t)b * TDEC + t) * DMODEL;
  float* dst = HID + (size_t)i * DMODEL;
  for (int c = threadIdx.x; c < DMODEL; c += blockDim.x) dst[c] = src[c];
}

// ---------------------------------------------------------------------------
// Orchestration
// ---------------------------------------------------------------------------
extern "C" void kernel_launch(void* const* d_in, const int* in_sizes, int n_in,
                              void* d_out, int out_size, void* d_ws, size_t ws_size,
                              hipStream_t stream) {
  (void)in_sizes; (void)n_in; (void)out_size; (void)ws_size;

  const float* images = (const float*)d_in[0];
  const int*   ids    = (const int*)d_in[1];
  // d_in[2] = attention_mask (unused, matching reference)
  const float* states = (const float*)d_in[3];

  // params: jax tree-flatten order (dict keys sorted alphabetically)
  const float* act_b1   = (const float*)d_in[4];
  const float* act_b2   = (const float*)d_in[5];
  const float* act_w1   = (const float*)d_in[6];
  const float* act_w2   = (const float*)d_in[7];
  const int DEC0 = 8;                    // 6 blocks x 16 tensors
  const float* lm_b     = (const float*)d_in[104];
  const float* lm_w     = (const float*)d_in[105];
  const float* ln_f_b   = (const float*)d_in[106];
  const float* ln_f_g   = (const float*)d_in[107];
  const float* patch_b  = (const float*)d_in[108];
  const float* patch_w  = (const float*)d_in[109];
  const float* pos_emb  = (const float*)d_in[110];
  const float* proj_b   = (const float*)d_in[111];
  const float* proj_w   = (const float*)d_in[112];
  const float* st_b1    = (const float*)d_in[113];
  const float* st_b2    = (const float*)d_in[114];
  const float* st_w1    = (const float*)d_in[115];
  const float* st_w2    = (const float*)d_in[116];
  const float* tok_emb  = (const float*)d_in[117];
  const int VIT0 = 118;                  // 4 blocks x 16 tensors
  const float* vit_ln_b = (const float*)d_in[182];
  const float* vit_ln_g = (const float*)d_in[183];
  const float* vit_pos  = (const float*)d_in[184];
  // block-dict sorted key index: 0:b1 1:b2 2:bk 3:bo 4:bq 5:bv 6:ln1_b 7:ln1_g
  //                              8:ln2_b 9:ln2_g 10:w1 11:w2 12:wk 13:wo 14:wq 15:wv

  // workspace layout (floats); total ~22.4M floats (~90 MB, fits global L2)
  float* ws = (float*)d_ws;
  const size_t ROWS = (size_t)BATCH * TDEC * DMODEL;   // 2,174,976
  float* Hb  = ws;                    // hidden (ViT x for phase 1)
  float* XN  = Hb  + ROWS;            // layernorm out
  float* Qb  = XN  + ROWS;
  float* Kb  = Qb  + ROWS;
  float* Vb  = Kb  + ROWS;
  float* AT  = Vb  + ROWS;            // attention out
  float* FF  = AT  + ROWS;            // FFN hidden (2832*3072 = 4*ROWS)
  float* XP  = FF  + 4 * ROWS;        // patches / img_proj (784*768)
  float* STb = XP  + (size_t)784 * DMODEL;
  float* SEb = STb + 4 * DMODEL;
  float* HID = SEb + 4 * DMODEL;
  float* A1  = HID + 8 * DMODEL;
  int* pos_state = (int*)(A1 + 8 * DMODEL);
  int* pos_act   = pos_state + BATCH * 2;

  auto gemm = [&](const float* A, const float* W, const float* bias, float* C,
                  const float* res, const float* aux, int M, int N, int K,
                  int auxmod, int fuse_gelu) {
    dim3 g((N + 127) / 128, (M + 63) / 64);
    gemm_wmma<<<g, 256, 0, stream>>>(A, W, bias, C, res, aux, M, N, K,
                                     auxmod, fuse_gelu);
  };

  // ============================ ViT (784 rows) ============================
  patchify<<<(784 * 768 + 255) / 256, 256, 0, stream>>>(images, XP);
  gemm(XP, patch_w, patch_b, Hb, nullptr, vit_pos, 784, DMODEL, DMODEL, TVIT, 0);

  for (int L = 0; L < 4; ++L) {
    auto P = [&](int j) { return (const float*)d_in[VIT0 + L * 16 + j]; };
    layernorm_row<<<784, 32, 0, stream>>>(Hb, XN, P(7), P(6), DMODEL);
    gemm(XN, P(14), P(4), Qb, nullptr, nullptr, 784, DMODEL, DMODEL, 1, 0);
    gemm(XN, P(12), P(2), Kb, nullptr, nullptr, 784, DMODEL, DMODEL, 1, 0);
    gemm(XN, P(15), P(5), Vb, nullptr, nullptr, 784, DMODEL, DMODEL, 1, 0);
    attn_wmma<<<dim3((TVIT + 15) / 16, NHEAD, 4), 32, 0, stream>>>(
        Qb, Kb, Vb, AT, TVIT, 0);
    gemm(AT, P(13), P(3), Hb, Hb, nullptr, 784, DMODEL, DMODEL, 1, 0);
    layernorm_row<<<784, 32, 0, stream>>>(Hb, XN, P(9), P(8), DMODEL);
    gemm(XN, P(10), P(0), FF, nullptr, nullptr, 784, FFDIM, DMODEL, 1, 1);
    gemm(FF, P(11), P(1), Hb, Hb, nullptr, 784, DMODEL, FFDIM, 1, 0);
  }
  layernorm_row<<<784, 32, 0, stream>>>(Hb, XN, vit_ln_g, vit_ln_b, DMODEL);
  gemm(XN, proj_w, proj_b, XP, nullptr, nullptr, 784, DMODEL, DMODEL, 1, 0);

  // ========================= state encoder (tiny) =========================
  small_gemm<<<(4 * DMODEL + 255) / 256, 256, 0, stream>>>(
      states, st_w1, st_b1, STb, 4, DMODEL, 16, 1);
  small_gemm<<<(4 * DMODEL + 255) / 256, 256, 0, stream>>>(
      STb, st_w2, st_b2, SEb, 4, DMODEL, DMODEL, 0);

  // ======================= assemble decoder input ========================
  find_positions<<<1, 32, 0, stream>>>(ids, pos_state, pos_act);
  embed_text<<<BATCH * SEQ, 256, 0, stream>>>(ids, tok_emb, pos_emb, Hb);
  assemble_img<<<784, 256, 0, stream>>>(XP, pos_emb, Hb);
  scatter_state<<<BATCH * 2, 256, 0, stream>>>(SEb, pos_emb, pos_state, Hb);

  // ========================= decoder (2832 rows) =========================
  const int M = BATCH * TDEC;   // 2832
  for (int L = 0; L < 6; ++L) {
    auto P = [&](int j) { return (const float*)d_in[DEC0 + L * 16 + j]; };
    layernorm_row<<<M, 32, 0, stream>>>(Hb, XN, P(7), P(6), DMODEL);
    gemm(XN, P(14), P(4), Qb, nullptr, nullptr, M, DMODEL, DMODEL, 1, 0);
    gemm(XN, P(12), P(2), Kb, nullptr, nullptr, M, DMODEL, DMODEL, 1, 0);
    gemm(XN, P(15), P(5), Vb, nullptr, nullptr, M, DMODEL, DMODEL, 1, 0);
    attn_wmma<<<dim3((TDEC + 15) / 16, NHEAD, BATCH), 32, 0, stream>>>(
        Qb, Kb, Vb, AT, TDEC, 1);
    gemm(AT, P(13), P(3), Hb, Hb, nullptr, M, DMODEL, DMODEL, 1, 0);
    layernorm_row<<<M, 32, 0, stream>>>(Hb, XN, P(9), P(8), DMODEL);
    gemm(XN, P(10), P(0), FF, nullptr, nullptr, M, FFDIM, DMODEL, 1, 1);
    gemm(FF, P(11), P(1), Hb, Hb, nullptr, M, DMODEL, FFDIM, 1, 0);
  }
  layernorm_row<<<M, 32, 0, stream>>>(Hb, XN, ln_f_g, ln_f_b, DMODEL);

  // LM head: the single biggest GEMM/memory op -> straight into d_out
  gemm(XN, lm_w, lm_b, (float*)d_out, nullptr, nullptr, M, VOCAB, DMODEL, 1, 0);

  // ========================= action decoder (tiny) =======================
  gather_hid<<<BATCH * 4, 256, 0, stream>>>(XN, pos_act, HID);
  small_gemm<<<(8 * DMODEL + 255) / 256, 256, 0, stream>>>(
      HID, act_w1, act_b1, A1, 8, DMODEL, DMODEL, 1);
  float* act_out = (float*)d_out + (size_t)M * VOCAB;   // after logits
  small_gemm<<<(8 * 56 + 255) / 256, 256, 0, stream>>>(
      A1, act_w2, act_b2, act_out, 8, 56, DMODEL, 0);
}
